// MutiCaptionGenerator_44650480009871
// MI455X (gfx1250) — compile-verified, matrix operand order checked by reading, not run on hardware
//
#include <hip/hip_runtime.h>
#include <hip/hip_bf16.h>
#include <math.h>

// ---------------- problem constants ----------------
#define BB   64
#define SS   256
#define EE   1024
#define HH   16
#define DH   64
#define NLAY 2
#define FFD  4096
#define MM   (BB * SS)          // 16384 rows
#define E3   (3 * EE)           // 3072

typedef _Float16 f16;
typedef _Float16 v8h  __attribute__((ext_vector_type(8)));
typedef _Float16 v16h __attribute__((ext_vector_type(16)));
typedef float    v8f  __attribute__((ext_vector_type(8)));
typedef __attribute__((address_space(3))) f16 lds_f16;

// 16-bit A/B fragment layout (CDNA5 ISA 7.12.2): lane holds one row (A: m=lane&15)
// or one column (B: n=lane&15); element e maps to k = (e&7) + (lane>>4)*8 + (e>>3)*16.
// => elements 0..7 contiguous at k0 = half*8, elements 8..15 at k0+16.
__device__ __forceinline__ v16h load_frag(const f16* rowptr, int kh) {
    v8h lo = *(const v8h*)(rowptr + kh * 8);
    v8h hi = *(const v8h*)(rowptr + kh * 8 + 16);
    return __builtin_shufflevector(lo, hi, 0,1,2,3,4,5,6,7,8,9,10,11,12,13,14,15);
}

// gfx1250 async copy: global -> LDS, 16 bytes per lane, tracked by ASYNCcnt.
__device__ __forceinline__ void async_b128(f16* lds_dst, const f16* gsrc) {
    unsigned loff = (unsigned)(size_t)(lds_f16*)lds_dst;
    unsigned long long ga = (unsigned long long)(size_t)gsrc;
    asm volatile("global_load_async_to_lds_b128 %0, %1, off"
                 :: "v"(loff), "v"(ga) : "memory");
}
__device__ __forceinline__ void wait_async0() {
    asm volatile("s_wait_asynccnt 0x0" ::: "memory");
}

__device__ __forceinline__ float gelu_new(float x) {
    return 0.5f * x * (1.0f + tanhf(0.7978845608028654f * (x + 0.044715f * x * x * x)));
}

// ---------------- elementwise / conversion kernels ----------------
__global__ __launch_bounds__(256) void add_wpe_kernel(
        const float* __restrict__ sh, const float* __restrict__ wpe,
        float* __restrict__ h) {
    size_t i = (size_t)blockIdx.x * 256 + threadIdx.x;
    h[i] = sh[i] + wpe[i & ((size_t)SS * EE - 1)];
}

// f32 [K,N] row-major -> f16 [N,K] row-major (transpose + downconvert)
__global__ __launch_bounds__(256) void convtr_kernel(
        const float* __restrict__ in, f16* __restrict__ out, int K, int N) {
    size_t i = (size_t)blockIdx.x * 256 + threadIdx.x;
    if (i >= (size_t)K * N) return;
    int k = (int)(i / N);
    int n = (int)(i % N);
    out[(size_t)n * K + k] = (f16)in[i];
}

// ---------------- layernorm ----------------
template<bool OUT32>
__global__ __launch_bounds__(256) void ln_kernel(
        const float* __restrict__ in, const float* __restrict__ g,
        const float* __restrict__ b, f16* __restrict__ o16,
        float* __restrict__ o32) {
    __shared__ float r1[256];
    __shared__ float r2[256];
    const int row = blockIdx.x, tid = threadIdx.x;
    const float* x = in + (size_t)row * EE;
    float v[4], s = 0.f, q = 0.f;
#pragma unroll
    for (int i = 0; i < 4; ++i) {
        v[i] = x[tid + i * 256];
        s += v[i];
        q += v[i] * v[i];
    }
    r1[tid] = s; r2[tid] = q;
    __syncthreads();
    for (int st = 128; st > 0; st >>= 1) {
        if (tid < st) { r1[tid] += r1[tid + st]; r2[tid] += r2[tid + st]; }
        __syncthreads();
    }
    const float mean = r1[0] * (1.f / EE);
    const float var  = r2[0] * (1.f / EE) - mean * mean;
    const float rstd = rsqrtf(var + 1e-5f);
#pragma unroll
    for (int i = 0; i < 4; ++i) {
        int c = tid + i * 256;
        float y = (v[i] - mean) * rstd * g[c] + b[c];
        if (OUT32) o32[(size_t)row * EE + c] = y;
        else       o16[(size_t)row * EE + c] = (f16)y;
    }
}

// ---------------- generic WMMA GEMM ----------------
// C[M,N] = A[M,K](f16) * Bt[N,K]^T(f16) + bias ; block tile 128x128, 8 waves,
// each wave 16 rows x 128 cols (8 accumulators). W tile double-buffered in LDS,
// filled with gfx1250 async global->LDS copies overlapped with WMMA compute.
// MODE 0: store f16     MODE 1: gelu, store f16     MODE 2: +resid, store f32
template<int MODE>
__global__ __launch_bounds__(256, 2) void gemm_kernel(
        const f16* __restrict__ A, const f16* __restrict__ Bt,
        const float* __restrict__ bias, const float* __restrict__ resid,
        f16* __restrict__ out16, float* __restrict__ out32,
        int Ndim, int Kdim) {
    __shared__ f16 bt[2][128 * 32];      // double-buffered B tile: 128 cols x 32 k
    const int tid  = threadIdx.x;
    const int wave = tid >> 5;
    const int lane = tid & 31;
    const int kh   = lane >> 4;
    const int l15  = lane & 15;
    const int n0   = blockIdx.x * 128;
    const int m0   = blockIdx.y * 128 + wave * 16;

    const f16* arow = A + (size_t)(m0 + l15) * Kdim;   // A row for this lane

    v8f acc[8];
#pragma unroll
    for (int i = 0; i < 8; ++i)
#pragma unroll
        for (int j = 0; j < 8; ++j) acc[i][j] = 0.f;

    // staging map: thread -> col nl (0..127), k-offset 0 or 16; 2x b128 each
    const int snl = tid >> 1;
    const int sk  = (tid & 1) << 4;
    const f16* brow = Bt + (size_t)(n0 + snl) * Kdim + sk;
    f16* sdst = &bt[0][0] + (snl * 32 + sk);

    const int kchunks = Kdim >> 5;

    // prologue: fill buffer 0
    async_b128(sdst, brow);
    async_b128(sdst + 8, brow + 8);
    wait_async0();
    __syncthreads();

    for (int kc = 0; kc < kchunks; ++kc) {
        const int cur = kc & 1;
        // prefetch next W tile into the other buffer (overlaps the WMMAs below)
        if (kc + 1 < kchunks) {
            const f16* src = brow + (size_t)(kc + 1) * 32;
            f16* dst = &bt[cur ^ 1][0] + (snl * 32 + sk);
            async_b128(dst, src);
            async_b128(dst + 8, src + 8);
        }

        v16h af = load_frag(arow + kc * 32, kh);
#pragma unroll
        for (int nt = 0; nt < 8; ++nt) {
            v16h bf = load_frag(&bt[cur][(nt * 16 + l15) * 32], kh);
            acc[nt] = __builtin_amdgcn_wmma_f32_16x16x32_f16(
                false, af, false, bf, (short)0, acc[nt], false, false);
        }

        wait_async0();
        __syncthreads();
    }

    // epilogue: lane holds col n=l15, rows m = kh*8 + j
#pragma unroll
    for (int nt = 0; nt < 8; ++nt) {
        const int n = n0 + nt * 16 + l15;
        const float bv = bias[n];
#pragma unroll
        for (int j = 0; j < 8; ++j) {
            const size_t mrow = (size_t)(m0 + kh * 8 + j);
            float v = acc[nt][j] + bv;
            if (MODE == 0) {
                out16[mrow * Ndim + n] = (f16)v;
            } else if (MODE == 1) {
                out16[mrow * Ndim + n] = (f16)gelu_new(v);
            } else {
                out32[mrow * Ndim + n] = resid[mrow * Ndim + n] + v;
            }
        }
    }
}

// ---------------- attention (one wave per (b, head, 16-query tile)) ----------------
__global__ __launch_bounds__(32) void attn_kernel(
        const f16* __restrict__ qkv, f16* __restrict__ a16) {
    __shared__ float sc[16 * 256];   // scores, then exp values
    __shared__ f16   pf[16 * 256];   // softmax probs (f16)
    __shared__ f16   vt[32 * 64];    // staged V chunk

    const int lane = threadIdx.x;
    const int kh   = lane >> 4;
    const int l15  = lane & 15;
    const int q0   = blockIdx.x * 16;
    const int hh   = blockIdx.y;
    const int b    = blockIdx.z;
    const size_t rowbase = (size_t)b * SS;

    // Q fragments (head slice of this lane's query row), D=64 -> two 32-k chunks
    const f16* qrow = qkv + (rowbase + q0 + l15) * E3 + hh * DH;
    v16h aq0 = load_frag(qrow, kh);
    v16h aq1 = load_frag(qrow + 32, kh);

    // ---- scores = Q K^T, masked + scaled, into LDS ----
    for (int kt = 0; kt < 16; ++kt) {
        const f16* krow = qkv + (rowbase + kt * 16 + l15) * E3 + EE + hh * DH;
        v16h bk0 = load_frag(krow, kh);
        v16h bk1 = load_frag(krow + 32, kh);
        v8f c;
#pragma unroll
        for (int j = 0; j < 8; ++j) c[j] = 0.f;
        c = __builtin_amdgcn_wmma_f32_16x16x32_f16(false, aq0, false, bk0,
                                                   (short)0, c, false, false);
        c = __builtin_amdgcn_wmma_f32_16x16x32_f16(false, aq1, false, bk1,
                                                   (short)0, c, false, false);
        const int n = kt * 16 + l15;
#pragma unroll
        for (int j = 0; j < 8; ++j) {
            const int mrow = kh * 8 + j;
            const int qg = q0 + mrow;
            sc[mrow * 256 + n] = (n <= qg) ? c[j] * 0.125f : -10000.0f;
        }
    }
    __syncthreads();

    // ---- row softmax over 256 keys (threads 0..15, one row each) ----
    if (lane < 16) {
        float* r = &sc[lane * 256];
        float mx = -3.4e38f;
        for (int i = 0; i < 256; ++i) mx = fmaxf(mx, r[i]);
        float s = 0.f;
        for (int i = 0; i < 256; ++i) {
            float e = expf(r[i] - mx);
            r[i] = e;
            s += e;
        }
        const float inv = 1.f / s;
        f16* p = &pf[lane * 256];
        for (int i = 0; i < 256; ++i) p[i] = (f16)(r[i] * inv);
    }
    __syncthreads();

    // ---- out = P V  (8 chunks of 32 keys, V staged via async global->LDS) ----
    v8f acc[4];
#pragma unroll
    for (int i = 0; i < 4; ++i)
#pragma unroll
        for (int j = 0; j < 8; ++j) acc[i][j] = 0.f;

    for (int kc = 0; kc < 8; ++kc) {
        __syncthreads();
        {   // stage 32 x 64 V chunk: each lane copies one key row (8x async b128)
            const f16* vrow = qkv + (rowbase + kc * 32 + lane) * E3 + 2 * EE + hh * DH;
#pragma unroll
            for (int i = 0; i < 8; ++i)
                async_b128(&vt[lane * 64 + i * 8], vrow + i * 8);
        }
        wait_async0();
        __syncthreads();

        v16h ap = load_frag(&pf[l15 * 256 + kc * 32], kh);
#pragma unroll
        for (int nt = 0; nt < 4; ++nt) {
            v16h bv;
#pragma unroll
            for (int e = 0; e < 16; ++e) {
                const int k = (e & 7) + kh * 8 + ((e >> 3) << 4);
                bv[e] = vt[k * 64 + nt * 16 + l15];
            }
            acc[nt] = __builtin_amdgcn_wmma_f32_16x16x32_f16(
                false, ap, false, bv, (short)0, acc[nt], false, false);
        }
    }

#pragma unroll
    for (int nt = 0; nt < 4; ++nt)
#pragma unroll
        for (int j = 0; j < 8; ++j) {
            const int mrow = kh * 8 + j;
            a16[(rowbase + q0 + mrow) * EE + hh * DH + nt * 16 + l15] =
                (f16)acc[nt][j];
        }
}

// ---------------- loss ----------------
__global__ void zero_loss_kernel(float* loss) { loss[0] = 0.f; }

__global__ __launch_bounds__(256) void loss_kernel(
        const float* __restrict__ hout, const float* __restrict__ sh,
        float* __restrict__ loss) {
    __shared__ float red[256];
    const int s = blockIdx.x;          // 0..254
    const int tid = threadIdx.x;
    float acc = 0.f;
#pragma unroll
    for (int i = 0; i < 4; ++i) {
        const int c = tid + i * 256;
        const float d = hout[(size_t)s * EE + c] - sh[(size_t)(s + 1) * EE + c];
        acc += d * d;
    }
    red[tid] = acc;
    __syncthreads();
    for (int st = 128; st > 0; st >>= 1) {
        if (tid < st) red[tid] += red[tid + st];
        __syncthreads();
    }
    if (tid == 0) atomicAdd(loss, red[0] * (1.0f / ((SS - 1) * (float)EE)));
}

// ---------------- host orchestration ----------------
static inline size_t align256(size_t x) { return (x + 255) & ~(size_t)255; }

extern "C" void kernel_launch(void* const* d_in, const int* in_sizes, int n_in,
                              void* d_out, int out_size, void* d_ws, size_t ws_size,
                              hipStream_t stream) {
    (void)in_sizes; (void)n_in; (void)out_size; (void)ws_size;
    const float* sh      = (const float*)d_in[0];
    const float* wpe     = (const float*)d_in[1];
    const float* ln1_g   = (const float*)d_in[2];
    const float* ln1_b   = (const float*)d_in[3];
    const float* attn_w  = (const float*)d_in[4];
    const float* attn_b  = (const float*)d_in[5];
    const float* aproj_w = (const float*)d_in[6];
    const float* aproj_b = (const float*)d_in[7];
    const float* ln2_g   = (const float*)d_in[8];
    const float* ln2_b   = (const float*)d_in[9];
    const float* fc_w    = (const float*)d_in[10];
    const float* fc_b    = (const float*)d_in[11];
    const float* mproj_w = (const float*)d_in[12];
    const float* mproj_b = (const float*)d_in[13];
    const float* lnf_g   = (const float*)d_in[14];
    const float* lnf_b   = (const float*)d_in[15];

    char* ws = (char*)d_ws;
    size_t off = 0;
    auto take = [&](size_t bytes) { char* p = ws + off; off += align256(bytes); return p; };

    float* h   = (float*)take((size_t)MM * EE * 4);     // residual stream (f32)
    f16*   x16 = (f16*)  take((size_t)MM * EE * 2);     // LN output (f16)
    f16*   a16 = (f16*)  take((size_t)MM * EE * 2);     // attention output (f16)
    f16*   big = (f16*)  take((size_t)MM * FFD * 2);    // qkv16 (96MB) / ff16 (128MB) union
    // transposed f16 weights, per-layer blocks
    const size_t wqkvN = (size_t)E3 * EE, wprjN = (size_t)EE * EE;
    const size_t wfcN  = (size_t)FFD * EE, wmlpN = (size_t)EE * FFD;
    f16* wT = (f16*)take(NLAY * (wqkvN + wprjN + wfcN + wmlpN) * 2);

    float* hout = (float*)d_out;
    float* loss = (float*)d_out + (size_t)MM * EE;

    // --- weight conversion (f32 -> transposed f16) ---
    for (int l = 0; l < NLAY; ++l) {
        f16* base = wT + l * (wqkvN + wprjN + wfcN + wmlpN);
        f16* wqkvT = base;
        f16* wprjT = wqkvT + wqkvN;
        f16* wfcT  = wprjT + wprjN;
        f16* wmlpT = wfcT + wfcN;
        convtr_kernel<<<(int)((wqkvN + 255) / 256), 256, 0, stream>>>(
            attn_w + (size_t)l * EE * E3, wqkvT, EE, E3);
        convtr_kernel<<<(int)((wprjN + 255) / 256), 256, 0, stream>>>(
            aproj_w + (size_t)l * EE * EE, wprjT, EE, EE);
        convtr_kernel<<<(int)((wfcN + 255) / 256), 256, 0, stream>>>(
            fc_w + (size_t)l * EE * FFD, wfcT, EE, FFD);
        convtr_kernel<<<(int)((wmlpN + 255) / 256), 256, 0, stream>>>(
            mproj_w + (size_t)l * FFD * EE, wmlpT, FFD, EE);
    }

    // --- embeddings ---
    add_wpe_kernel<<<(int)((size_t)MM * EE / 256), 256, 0, stream>>>(sh, wpe, h);

    // --- transformer layers ---
    for (int l = 0; l < NLAY; ++l) {
        f16* base = wT + l * (wqkvN + wprjN + wfcN + wmlpN);
        f16* wqkvT = base;
        f16* wprjT = wqkvT + wqkvN;
        f16* wfcT  = wprjT + wprjN;
        f16* wmlpT = wfcT + wfcN;

        ln_kernel<false><<<MM, 256, 0, stream>>>(h, ln1_g + l * EE, ln1_b + l * EE,
                                                 x16, nullptr);
        gemm_kernel<0><<<dim3(E3 / 128, MM / 128), 256, 0, stream>>>(
            x16, wqkvT, attn_b + l * E3, nullptr, big, nullptr, E3, EE);
        attn_kernel<<<dim3(SS / 16, HH, BB), 32, 0, stream>>>(big, a16);
        gemm_kernel<2><<<dim3(EE / 128, MM / 128), 256, 0, stream>>>(
            a16, wprjT, aproj_b + l * EE, h, nullptr, h, EE, EE);

        ln_kernel<false><<<MM, 256, 0, stream>>>(h, ln2_g + l * EE, ln2_b + l * EE,
                                                 x16, nullptr);
        gemm_kernel<1><<<dim3(FFD / 128, MM / 128), 256, 0, stream>>>(
            x16, wfcT, fc_b + l * FFD, nullptr, big, nullptr, FFD, EE);
        gemm_kernel<2><<<dim3(EE / 128, MM / 128), 256, 0, stream>>>(
            big, wmlpT, mproj_b + l * EE, h, nullptr, h, EE, FFD);
    }

    // --- final LN -> d_out, then loss ---
    ln_kernel<true><<<MM, 256, 0, stream>>>(h, lnf_g, lnf_b, nullptr, hout);
    zero_loss_kernel<<<1, 1, 0, stream>>>(loss);
    loss_kernel<<<SS - 1, 256, 0, stream>>>(hout, sh, loss);
}
